// LeViT_UNet_192_63118839382167
// MI455X (gfx1250) — matrix operand order repaired and checked by reading, not compiled, not run on hardware
//
#include <hip/hip_runtime.h>
#include <cmath>

// ---------------------------------------------------------------------------
// Types for CDNA5 WMMA (gfx1250, wave32)
// ---------------------------------------------------------------------------
typedef _Float16 f16;
typedef __attribute__((ext_vector_type(16))) _Float16 v16h;
typedef __attribute__((ext_vector_type(8)))  float    v8f;

__device__ __forceinline__ float hswish_f(float v) {
    return v * fminf(fmaxf(v + 3.0f, 0.0f), 6.0f) * (1.0f / 6.0f);
}
__device__ __forceinline__ int iabs_i(int v) { return v < 0 ? -v : v; }

// ---------------------------------------------------------------------------
// f32 -> f16 bulk conversion (weights are static: converted once per call)
// ---------------------------------------------------------------------------
__global__ void f32_to_f16_k(const float* __restrict__ src, f16* __restrict__ dst, long n)
{
    for (long i = (long)blockIdx.x * blockDim.x + threadIdx.x; i < n;
         i += (long)gridDim.x * blockDim.x)
        dst[i] = (f16)src[i];
}

// ---------------------------------------------------------------------------
// Universal WMMA GEMM:  out[M,N] = act( A[M,K] * W[N,K]^T * scale + shift ) + residual
//   A is f32; W is pre-converted f16 ([N,K] row-major).
//   amode  : 0 = A row-major [M,K] (Linear); 1 = implicit im2col of NCHW (3x3, pad1, stride cs)
//   outmode: 0 = row-major [M,N] (== token-major); 1 = NCHW [B,N,Hout,Wout]
//   act    : 0 none, 1 hswish, 2 relu (after scale/shift, before residual)
// Block tile 64x64x32, 128 threads (4 waves), each wave = 2x2 WMMA 16x16x32 f16.
// Software-pipelined: double-buffered LDS; next tile's global loads are issued
// before the current tile's WMMAs so VMEM latency hides behind matrix math.
// ---------------------------------------------------------------------------
__global__ __launch_bounds__(128) void wmma_gemm(
    const float* __restrict__ A, const f16* __restrict__ Wt,
    const float* __restrict__ scale, const float* __restrict__ shift,
    const float* __restrict__ residual, float* __restrict__ out,
    int M, int N, int K, int amode, int outmode, int act,
    int Cin, int Hin, int Win, int Hout, int Wout, int cs)
{
    __shared__ f16 AsmAll[2 * 2560];   // 2 x (64 rows x 40 f16)
    __shared__ f16 BsmAll[2 * 2560];

    const int tid = threadIdx.x;
    const int m0 = blockIdx.x * 64, n0 = blockIdx.y * 64;
    const int wid = tid >> 5, lane = tid & 31;
    const int waveM = wid >> 1, waveN = wid & 1;
    const int half = lane >> 4, mr = lane & 15;
    const int hw = Hout * Wout;

    v8f acc[2][2];
#pragma unroll
    for (int i = 0; i < 2; i++)
#pragma unroll
        for (int j = 0; j < 2; j++)
#pragma unroll
            for (int r = 0; r < 8; r++) acc[i][j][r] = 0.0f;

    // im2col: per-thread row-walk start (decoded ONCE, walked incrementally)
    int bw0 = 0, oyw0 = 0, oxw0 = 0;
    if (amode == 1) {
        int gm0 = m0 + (tid >> 5);
        bw0 = gm0 / hw;
        int rem = gm0 - bw0 * hw;
        oyw0 = rem / Wout;
        oxw0 = rem - oyw0 * Wout;
    }

    const bool kvec4 = (K & 3) == 0;
    const bool kvec8 = (K & 7) == 0;
    const int ksteps = (K + 31) >> 5;

    // ---- register staging for the software pipeline ----
    uint2 aLin[4];     // linear-A: 4 chunks of 4 f16
    f16   aConv[16];   // im2col-A: 16 scalars (fixed column, 16 rows)
    uint4 bR[2];       // B: 2 chunks of 8 f16

    auto stageLoad = [&](int k0) {
        // ---- W tile: 64 rows x 32 f16, 2 chunks of 8 per thread ----
#pragma unroll
        for (int u = 0; u < 2; u++) {
            int q = tid + (u << 7);
            int r = q >> 2, c8 = (q & 3) << 3;
            int gn = n0 + r, kk = k0 + c8;
            union { f16 h[8]; uint4 u4; } t;
            if (kvec8 && gn < N && kk + 7 < K) {
                t.u4 = *(const uint4*)(Wt + (size_t)gn * K + kk);
            } else {
#pragma unroll
                for (int e = 0; e < 8; e++)
                    t.h[e] = (gn < N && kk + e < K) ? Wt[(size_t)gn * K + kk + e] : (f16)0.f;
            }
            bR[u] = t.u4;
        }
        // ---- A tile ----
        if (amode == 0) {
#pragma unroll
            for (int u = 0; u < 4; u++) {
                int q = tid + (u << 7);
                int r = q >> 3, c4 = (q & 7) << 2;
                int gm = m0 + r, kk = k0 + c4;
                union { f16 h[4]; uint2 u2; } o;
                if (kvec4 && gm < M && kk + 3 < K) {
                    float4 f = *(const float4*)(A + (size_t)gm * K + kk);
                    o.h[0] = (f16)f.x; o.h[1] = (f16)f.y;
                    o.h[2] = (f16)f.z; o.h[3] = (f16)f.w;
                } else {
#pragma unroll
                    for (int e = 0; e < 4; e++)
                        o.h[e] = (gm < M && kk + e < K) ? (f16)A[(size_t)gm * K + kk + e]
                                                        : (f16)0.f;
                }
                aLin[u] = o.u2;
            }
        } else {
            // fixed column c; (ci,ky,kx) decoded once per K-step; row walk with
            // carry-based (b,oy,ox) updates -- no divides in the hot loop.
            const int c = tid & 31, r0w = tid >> 5;
            const int gk = k0 + c;
            const bool kok = gk < K;
            int ci = 0, ky = 0, kx = 0;
            if (kok) {
                ci = gk / 9;
                int r9 = gk - ci * 9;
                ky = r9 / 3; kx = r9 - ky * 3;
            }
            int gm = m0 + r0w, b = bw0, oy = oyw0, ox = oxw0;
            (void)r0w;
#pragma unroll 4
            for (int nn = 0; nn < 16; nn++) {
                float av = 0.0f;
                if (gm < M && kok) {
                    int iy = oy * cs + ky - 1, ix = ox * cs + kx - 1;
                    if ((unsigned)iy < (unsigned)Hin && (unsigned)ix < (unsigned)Win)
                        av = A[(((size_t)b * Cin + ci) * Hin + iy) * Win + ix];
                }
                aConv[nn] = (f16)av;
                gm += 4; ox += 4;
                if (ox >= Wout) {
                    ox -= Wout; oy++;
                    if (oy >= Hout) { oy = 0; b++; }
                }
            }
        }
    };

    auto stageStore = [&](f16* As, f16* Bs) {
#pragma unroll
        for (int u = 0; u < 2; u++) {
            int q = tid + (u << 7);
            int r = q >> 2, c8 = (q & 3) << 3;
            *(uint4*)&Bs[r * 40 + c8] = bR[u];
        }
        if (amode == 0) {
#pragma unroll
            for (int u = 0; u < 4; u++) {
                int q = tid + (u << 7);
                int r = q >> 3, c4 = (q & 7) << 2;
                *(uint2*)&As[r * 40 + c4] = aLin[u];
            }
        } else {
            const int c = tid & 31, r0w = tid >> 5;
#pragma unroll 4
            for (int nn = 0; nn < 16; nn++)
                As[(r0w + 4 * nn) * 40 + c] = aConv[nn];
        }
    };

    // ---- pipeline prologue: stage tile 0 ----
    stageLoad(0);
    stageStore(&AsmAll[0], &BsmAll[0]);
    __syncthreads();

    for (int kt = 0; kt < ksteps; kt++) {
        f16* Ac = &AsmAll[(kt & 1) * 2560];
        f16* Bc = &BsmAll[(kt & 1) * 2560];
        const bool more = (kt + 1) < ksteps;

        // issue next tile's global loads first (latency overlaps WMMAs below)
        if (more) stageLoad((kt + 1) << 5);

        // ---- build fragments per ISA 7.12.2 16-bit layouts, 4 WMMAs ----
        v16h afr[2], bfr[2];
#pragma unroll
        for (int i = 0; i < 2; i++) {
            int row = waveM * 32 + i * 16 + mr;
            const unsigned int* pa = (const unsigned int*)&Ac[row * 40 + 8 * half];
            const unsigned int* pb = (const unsigned int*)&Ac[row * 40 + 16 + 8 * half];
            union { unsigned int d[8]; v16h v; } u;
            u.d[0] = pa[0]; u.d[1] = pa[1]; u.d[2] = pa[2]; u.d[3] = pa[3];
            u.d[4] = pb[0]; u.d[5] = pb[1]; u.d[6] = pb[2]; u.d[7] = pb[3];
            afr[i] = u.v;
        }
#pragma unroll
        for (int j = 0; j < 2; j++) {
            int row = waveN * 32 + j * 16 + mr;
            const unsigned int* p = (const unsigned int*)&Bc[row * 40 + 8 * half];
            const unsigned int* p2 = (const unsigned int*)&Bc[row * 40 + 16 + 8 * half];
            union { unsigned int d[8]; v16h v; } u;
#pragma unroll
            for (int q = 0; q < 4; q++) { u.d[q] = p[q]; u.d[q + 4] = p2[q]; }
            bfr[j] = u.v;
        }
#pragma unroll
        for (int i = 0; i < 2; i++)
#pragma unroll
            for (int j = 0; j < 2; j++)
                acc[i][j] = __builtin_amdgcn_wmma_f32_16x16x32_f16(
                    false, afr[i], false, bfr[j], (short)0, acc[i][j], false, false);

        // write staged registers into the other LDS buffer
        if (more)
            stageStore(&AsmAll[((kt + 1) & 1) * 2560], &BsmAll[((kt + 1) & 1) * 2560]);
        __syncthreads();
    }

    // ---- epilogue: BN fold, activation, residual, store ----
#pragma unroll
    for (int j = 0; j < 2; j++) {
        int gn = n0 + waveN * 32 + j * 16 + mr;
        if (gn >= N) continue;
        float sc = scale ? scale[gn] : 1.0f;
        float sh = shift ? shift[gn] : 0.0f;
#pragma unroll
        for (int i = 0; i < 2; i++) {
#pragma unroll
            for (int r = 0; r < 8; r++) {
                int gm = m0 + waveM * 32 + i * 16 + r + 8 * half;
                if (gm >= M) continue;
                float v = acc[i][j][r] * sc + sh;
                if (act == 1) v = hswish_f(v);
                else if (act == 2) v = fmaxf(v, 0.0f);
                if (residual) v += residual[(size_t)gm * N + gn];
                if (outmode == 0) out[(size_t)gm * N + gn] = v;
                else {
                    int b = gm / hw, rem = gm - b * hw;
                    out[((size_t)b * N + gn) * hw + rem] = v;
                }
            }
        }
    }
}

// ---------------------------------------------------------------------------
// Fused attention row kernel: one wave per (b, head, query-token), dk = 32.
// Bias index computed analytically: idx = |qs*yq - yk| * reskv + |qs*xq - xk|
// hswish applied to the output (it always feeds proj).
// ---------------------------------------------------------------------------
__global__ __launch_bounds__(256) void attn_kernel(
    const float* __restrict__ Q, int qRow, int qHead, int qOff,
    const float* __restrict__ KV, int kRow, int kHead, int kOff, int vOff,
    const float* __restrict__ ab, int n_off,
    float* __restrict__ out,
    int B, int Nq, int Nkv, int heads, int d, int resq, int reskv, int qs)
{
    __shared__ float wsm[8 * 200];
    const int tid = threadIdx.x, wv = tid >> 5, lane = tid & 31;
    long total = (long)B * heads * Nq;
    long gw = (long)blockIdx.x * 8 + wv;
    if (gw >= total) gw = total - 1;   // deterministic duplicate work for tail
    int n = (int)(gw % Nq);
    int h = (int)((gw / Nq) % heads);
    int b = (int)(gw / ((long)Nq * heads));

    const float* qp = Q + ((size_t)b * Nq + n) * qRow + (size_t)h * qHead + qOff;
    float q[32];
#pragma unroll
    for (int kk = 0; kk < 32; kk++) q[kk] = qp[kk];

    const int ny = n / resq, nx = n % resq;
    const float sm = 0.17677669529663689f;  // 32^-0.5
    float loc[7];
    float mx = -1e30f;
    int cnt = 0;
    for (int m = lane; m < Nkv; m += 32) {
        const float* kp = KV + ((size_t)b * Nkv + m) * kRow + (size_t)h * kHead + kOff;
        float dot = 0.0f;
#pragma unroll
        for (int kk = 0; kk < 32; kk++) dot += q[kk] * kp[kk];
        int my = m / reskv, mxx = m % reskv;
        int idx = iabs_i(qs * ny - my) * reskv + iabs_i(qs * nx - mxx);
        float l = dot * sm + ab[(size_t)h * n_off + idx];
        loc[cnt++] = l;
        mx = fmaxf(mx, l);
    }
    for (int s = 16; s > 0; s >>= 1) mx = fmaxf(mx, __shfl_xor(mx, s, 32));
    float sum = 0.0f;
    for (int t = 0; t < cnt; t++) { float e = __expf(loc[t] - mx); loc[t] = e; sum += e; }
    for (int s = 16; s > 0; s >>= 1) sum += __shfl_xor(sum, s, 32);
    float inv = 1.0f / sum;

    float* wrow = &wsm[wv * 200];
    { int t = 0; for (int m = lane; m < Nkv; m += 32) wrow[m] = loc[t++]; }
    __syncthreads();

    const float* vb = KV + (size_t)b * Nkv * kRow + (size_t)h * kHead + vOff;
    float accv[4] = {0, 0, 0, 0};
    const int nch = d >> 5;
    for (int m = 0; m < Nkv; m++) {
        float w = wrow[m];
        const float* vp = vb + (size_t)m * kRow;
        for (int t = 0; t < nch; t++) accv[t] += w * vp[t * 32 + lane];
    }
    float* op = out + ((size_t)b * Nq + n) * ((size_t)heads * d) + (size_t)h * d;
    for (int t = 0; t < nch; t++) op[t * 32 + lane] = hswish_f(accv[t] * inv);
}

// ---------------------------------------------------------------------------
// Elementwise helpers
// ---------------------------------------------------------------------------
__global__ void gather_sub_k(const float* __restrict__ src, float* __restrict__ dst,
                             int B, int res, int res_, int C)
{
    long total = (long)B * res_ * res_ * C;
    for (long i = (long)blockIdx.x * blockDim.x + threadIdx.x; i < total;
         i += (long)gridDim.x * blockDim.x) {
        int c = (int)(i % C);
        int j = (int)((i / C) % ((long)res_ * res_));
        int b = (int)(i / ((long)C * res_ * res_));
        int yq = j / res_, xq = j % res_;
        dst[i] = src[((size_t)b * res * res + (size_t)(2 * yq) * res + 2 * xq) * C + c];
    }
}

__global__ void tok2nchw_k(const float* __restrict__ t, float* __restrict__ dst,
                           int B, int C, int H, int W, int Ctot, int c0)
{
    long total = (long)B * C * H * W;
    for (long i = (long)blockIdx.x * blockDim.x + threadIdx.x; i < total;
         i += (long)gridDim.x * blockDim.x) {
        int x = (int)(i % W); long r = i / W;
        int y = (int)(r % H); r /= H;
        int c = (int)(r % C); int b = (int)(r / C);
        dst[(((size_t)b * Ctot + c0 + c) * H + y) * W + x] =
            t[((size_t)b * H * W + (size_t)y * W + x) * C + c];
    }
}

__global__ void copy_ch_k(const float* __restrict__ src, float* __restrict__ dst,
                          int B, int C, int H, int W, int Ctot, int c0)
{
    long total = (long)B * C * H * W;
    for (long i = (long)blockIdx.x * blockDim.x + threadIdx.x; i < total;
         i += (long)gridDim.x * blockDim.x) {
        int x = (int)(i % W); long r = i / W;
        int y = (int)(r % H); r /= H;
        int c = (int)(r % C); int b = (int)(r / C);
        dst[(((size_t)b * Ctot + c0 + c) * H + y) * W + x] =
            src[(((size_t)b * C + c) * H + y) * W + x];
    }
}

// bilinear, align_corners=True; src layout: 0=NCHW, 1=token-major [B,h*w,C]
__global__ void resize_bilinear_k(const float* __restrict__ src, float* __restrict__ dst,
                                  int B, int C, int h, int w, int oh, int ow,
                                  int Ctot, int c0, int srcTok)
{
    long total = (long)B * C * oh * ow;
    for (long i = (long)blockIdx.x * blockDim.x + threadIdx.x; i < total;
         i += (long)gridDim.x * blockDim.x) {
        int x = (int)(i % ow); long r = i / ow;
        int y = (int)(r % oh); r /= oh;
        int c = (int)(r % C); int b = (int)(r / C);
        double fy = (oh > 1) ? (double)y * (h - 1) / (oh - 1) : 0.0;
        double fx = (ow > 1) ? (double)x * (w - 1) / (ow - 1) : 0.0;
        int y0 = (int)fy; if (y0 > h - 2) y0 = h - 2; if (y0 < 0) y0 = 0;
        int x0 = (int)fx; if (x0 > w - 2) x0 = w - 2; if (x0 < 0) x0 = 0;
        int y1 = y0 + 1 < h ? y0 + 1 : h - 1;
        int x1 = x0 + 1 < w ? x0 + 1 : w - 1;
        float wy = (float)(fy - y0), wx = (float)(fx - x0);
        #define FETCH(yy, xx) (srcTok ? src[((size_t)b * h * w + (size_t)(yy) * w + (xx)) * C + c] \
                                      : src[(((size_t)b * C + c) * h + (yy)) * w + (xx)])
        float f00 = FETCH(y0, x0), f01 = FETCH(y0, x1);
        float f10 = FETCH(y1, x0), f11 = FETCH(y1, x1);
        #undef FETCH
        float v = (f00 * (1.f - wy) + f10 * wy) * (1.f - wx) +
                  (f01 * (1.f - wy) + f11 * wy) * wx;
        dst[(((size_t)b * Ctot + c0 + c) * oh + y) * ow + x] = v;
    }
}

// ---------------------------------------------------------------------------
// Host side
// ---------------------------------------------------------------------------
static inline int gs256(long total) {
    long b = (total + 255) / 256;
    if (b > 1048576) b = 1048576;
    if (b < 1) b = 1;
    return (int)b;
}

extern "C" void kernel_launch(void* const* d_in, const int* in_sizes, int n_in,
                              void* d_out, int out_size, void* d_ws, size_t ws_size,
                              hipStream_t stream)
{
    (void)n_in; (void)out_size; (void)ws_size;
    const float* const* FP = (const float* const*)d_in;
    const float* x = FP[0];
    float* ws = (float*)d_ws;
    const int B = 8;

    // ---- workspace layout (floats) ----
    const size_t SKIP0 = 0;                                  // [8,24,112,112]
    const size_t SKIP1 = SKIP0 + (size_t)8 * 24 * 112 * 112;
    const size_t SKIP2 = SKIP1 + (size_t)8 * 48 * 56 * 56;
    const size_t FEAT  = SKIP2 + (size_t)8 * 96 * 28 * 28;   // [8,864,14,14]
    const size_t D0    = FEAT + (size_t)8 * 864 * 196;       // max 8*152*112*112
    const size_t D1    = D0 + (size_t)8 * 152 * 112 * 112;   // max 8*64*224*224
    const size_t WF16  = D1 + (size_t)8 * 64 * 224 * 224;    // f16 weight pool (~6.8M fl)
    float* skip0 = ws + SKIP0;
    float* skip1 = ws + SKIP1;
    float* skip2 = ws + SKIP2;
    float* feat  = ws + FEAT;
    float* d0    = ws + D0;
    float* d1    = ws + D1;
    f16*   wpool = (f16*)(ws + WF16);
    // transformer scratch aliased into D1 (dead before D1's first decoder write)
    float* tokA = d1;
    float* tokB = d1 + 301056;
    float* ts   = d1 + 602112;               // qkv / kv / MLP hidden (max 1,505,280)
    float* atn  = d1 + 602112 + 1505280;     // attention output (max 301,056)
    float* xq   = atn + 301056;
    float* qb   = xq + 75264;

    // ---- pre-convert all GEMM weights to f16 (static data; tiny bandwidth) ----
    int list[80]; int nl = 0;
    list[nl++] = 1; list[nl++] = 4; list[nl++] = 7; list[nl++] = 10;          // stem
    auto addStage = [&](int base) {
        for (int k = 0; k < 4; k++) {
            int bb = base + 13 * k;
            list[nl++] = bb; list[nl++] = bb + 3; list[nl++] = bb + 7; list[nl++] = bb + 10;
        }
    };
    auto addDown = [&](int bb) {
        list[nl++] = bb; list[nl++] = bb + 3; list[nl++] = bb + 6;
        list[nl++] = bb + 10; list[nl++] = bb + 13;
    };
    addStage(13); addDown(65); addStage(81); addDown(133); addStage(149);
    for (int i = 0; i < 8; i++) list[nl++] = 201 + 3 * i;                     // decoder convs
    list[nl++] = 225;                                                        // head

    long offs[240];
    for (int i = 0; i < 240; i++) offs[i] = 0;
    long woff = 0;
    for (int ii = 0; ii < nl; ii++) {
        int idx = list[ii];
        offs[idx] = woff;
        long n = in_sizes[idx];
        f32_to_f16_k<<<gs256(n), 256, 0, stream>>>(FP[idx], wpool + woff, n);
        woff += (n + 7) & ~7L;   // 16-byte align each array for uint4 loads
    }

    auto WH = [&](int i) -> const f16* { return wpool + offs[i]; };
    auto gemm = [&](const float* A, int widx, const float* sc, const float* sh,
                    const float* res, float* out, int M, int N, int K,
                    int amode, int outmode, int act,
                    int Cin = 0, int Hin = 0, int Win = 0,
                    int Ho = 0, int Wo = 0, int cs = 1) {
        dim3 g((M + 63) / 64, (N + 63) / 64);
        wmma_gemm<<<g, 128, 0, stream>>>(A, WH(widx), sc, sh, res, out, M, N, K,
                                         amode, outmode, act, Cin, Hin, Win, Ho, Wo, cs);
    };

    // one transformer stage: depth x {attn + residual, MLP + residual}; tokens stay in tokA
    auto run_stage = [&](int base, int depth, int Ntok, int dim, int heads, int res) {
        const int C = heads * 128;  // heads*(2*32+64)
        const int M = B * Ntok;
        for (int blk = 0; blk < depth; blk++) {
            int bb = base + blk * 13;
            gemm(tokA, bb, FP[bb + 1], FP[bb + 2], nullptr, ts, M, C, dim, 0, 0, 0);
            long rows = (long)B * heads * Ntok;
            attn_kernel<<<(int)((rows + 7) / 8), 256, 0, stream>>>(
                ts, C, 128, 0, ts, C, 128, 32, 64,
                FP[bb + 6], res * res, atn, B, Ntok, Ntok, heads, 64, res, res, 1);
            gemm(atn, bb + 3, FP[bb + 4], FP[bb + 5], tokA, tokB, M, dim, heads * 64, 0, 0, 0);
            gemm(tokB, bb + 7, FP[bb + 8], FP[bb + 9], nullptr, ts, M, 2 * dim, dim, 0, 0, 1);
            gemm(ts, bb + 10, FP[bb + 11], FP[bb + 12], tokB, tokA, M, dim, 2 * dim, 0, 0, 0);
        }
    };

    // downsampling attention block (+ MLP); result ends in tokA
    auto run_down = [&](int bb, int Nin, int Nq, int dimIn, int dimOut,
                        int heads, int res, int res_) {
        const int kvC = heads * 160;  // heads*(32+128)
        gemm(tokA, bb, FP[bb + 1], FP[bb + 2], nullptr, ts, B * Nin, kvC, dimIn, 0, 0, 0);
        gather_sub_k<<<gs256((long)B * Nq * dimIn), 256, 0, stream>>>(tokA, xq, B, res, res_, dimIn);
        gemm(xq, bb + 3, FP[bb + 4], FP[bb + 5], nullptr, qb, B * Nq, heads * 32, dimIn, 0, 0, 0);
        long rows = (long)B * heads * Nq;
        attn_kernel<<<(int)((rows + 7) / 8), 256, 0, stream>>>(
            qb, heads * 32, 32, 0, ts, kvC, 160, 0, 32,
            FP[bb + 9], res * res, atn, B, Nq, Nin, heads, 128, res_, res, 2);
        gemm(atn, bb + 6, FP[bb + 7], FP[bb + 8], nullptr, tokB, B * Nq, dimOut, heads * 128, 0, 0, 0);
        gemm(tokB, bb + 10, FP[bb + 11], FP[bb + 12], nullptr, ts, B * Nq, 2 * dimOut, dimOut, 0, 0, 1);
        gemm(ts, bb + 13, FP[bb + 14], FP[bb + 15], tokB, tokA, B * Nq, dimOut, 2 * dimOut, 0, 0, 0);
    };

    // ---- stem: 4x conv3x3 s2 + BN (+hswish on first 3) via implicit-GEMM WMMA ----
    gemm(x,     1, FP[2], FP[3],   nullptr, skip0, B * 112 * 112, 24,  27,  1, 1, 1,  3, 224, 224, 112, 112, 2);
    gemm(skip0, 4, FP[5], FP[6],   nullptr, skip1, B * 56 * 56,   48,  216, 1, 1, 1, 24, 112, 112,  56,  56, 2);
    gemm(skip1, 7, FP[8], FP[9],   nullptr, skip2, B * 28 * 28,   96,  432, 1, 1, 1, 48,  56,  56,  28,  28, 2);
    gemm(skip2, 10, FP[11], FP[12], nullptr, tokA, B * 196,       192, 864, 1, 0, 0, 96,  28,  28,  14,  14, 2);

    // ---- transformer ----
    run_stage(13, 4, 196, 192, 3, 14);
    tok2nchw_k<<<gs256((long)B * 192 * 196), 256, 0, stream>>>(tokA, feat, B, 192, 14, 14, 864, 0);
    run_down(65, 196, 49, 192, 288, 6, 14, 7);
    run_stage(81, 4, 49, 288, 5, 7);
    resize_bilinear_k<<<gs256((long)B * 288 * 196), 256, 0, stream>>>(
        tokA, feat, B, 288, 7, 7, 14, 14, 864, 192, 1);
    run_down(133, 49, 16, 288, 384, 9, 7, 4);
    run_stage(149, 4, 16, 384, 6, 4);
    resize_bilinear_k<<<gs256((long)B * 384 * 196), 256, 0, stream>>>(
        tokA, feat, B, 384, 4, 4, 14, 14, 864, 480, 1);

    // ---- decoder ----
    resize_bilinear_k<<<gs256((long)B * 864 * 784), 256, 0, stream>>>(feat, d0, B, 864, 14, 14, 28, 28, 960, 0, 0);
    copy_ch_k<<<gs256((long)B * 96 * 784), 256, 0, stream>>>(skip2, d0, B, 96, 28, 28, 960, 864);
    gemm(d0, 201, FP[202], FP[203], nullptr, d1, B * 784, 256, 8640, 1, 1, 2, 960, 28, 28, 28, 28, 1);
    gemm(d1, 204, FP[205], FP[206], nullptr, d0, B * 784, 256, 2304, 1, 1, 2, 256, 28, 28, 28, 28, 1);

    resize_bilinear_k<<<gs256((long)B * 256 * 3136), 256, 0, stream>>>(d0, d1, B, 256, 28, 28, 56, 56, 304, 0, 0);
    copy_ch_k<<<gs256((long)B * 48 * 3136), 256, 0, stream>>>(skip1, d1, B, 48, 56, 56, 304, 256);
    gemm(d1, 207, FP[208], FP[209], nullptr, d0, B * 3136, 128, 2736, 1, 1, 2, 304, 56, 56, 56, 56, 1);
    gemm(d0, 210, FP[211], FP[212], nullptr, d1, B * 3136, 128, 1152, 1, 1, 2, 128, 56, 56, 56, 56, 1);

    resize_bilinear_k<<<gs256((long)B * 128 * 12544), 256, 0, stream>>>(d1, d0, B, 128, 56, 56, 112, 112, 152, 0, 0);
    copy_ch_k<<<gs256((long)B * 24 * 12544), 256, 0, stream>>>(skip0, d0, B, 24, 112, 112, 152, 128);
    gemm(d0, 213, FP[214], FP[215], nullptr, d1, B * 12544, 64, 1368, 1, 1, 2, 152, 112, 112, 112, 112, 1);
    gemm(d1, 216, FP[217], FP[218], nullptr, d0, B * 12544, 64, 576, 1, 1, 2, 64, 112, 112, 112, 112, 1);

    resize_bilinear_k<<<gs256((long)B * 64 * 50176), 256, 0, stream>>>(d0, d1, B, 64, 112, 112, 224, 224, 64, 0, 0);
    gemm(d1, 219, FP[220], FP[221], nullptr, d0, B * 50176, 32, 576, 1, 1, 2, 64, 224, 224, 224, 224, 1);
    gemm(d0, 222, FP[223], FP[224], nullptr, d1, B * 50176, 32, 288, 1, 1, 2, 32, 224, 224, 224, 224, 1);

    // ---- head: conv 32->9 with bias (scale=1, shift=b), no activation ----
    gemm(d1, 225, nullptr, FP[226], nullptr, (float*)d_out,
         B * 50176, 9, 288, 1, 1, 0, 32, 224, 224, 224, 224, 1);
}
